// EmformerAttention_34454227648705
// MI455X (gfx1250) — matrix operand order repaired and checked by reading, not compiled, not run on hardware
//
#include <hip/hip_runtime.h>
#include <hip/hip_bf16.h>
#include <stdint.h>

// ---------------- problem constants ----------------
#define CC 2048
#define RR 256
#define SS 16
#define MM 15
#define BB 2
#define DD 512
#define HH 8
#define QQ 2320            // R + C + S
#define KK 2319            // M + R + C
#define KP 2336            // K padded to 73*32 for 32-deep WMMA chunks
#define QTILES 145         // 2320/16
#define MIN_MASK (-1.0e8f)

typedef __attribute__((ext_vector_type(16))) __bf16 v16bf;
typedef __attribute__((ext_vector_type(8)))  float  v8f;

union ABu { v16bf v; uint4 q[2]; __bf16 h[16]; };

#define DEV static __device__ __forceinline__

DEV v8f wmma_bf16(v16bf a, v16bf b, v8f c) {
    // (neg_a, A, neg_b, B, c_mod, C, reuse_a, reuse_b) -> v_wmma_f32_16x16x32_bf16
    return __builtin_amdgcn_wmma_f32_16x16x32_bf16(false, a, false, b, (short)0, c, false, false);
}

// B fragment (32x16 bf16): lane holds col=lane&15, K chunks at half*8 and 16+half*8
DEV v16bf loadB_bf16(const __bf16* col_chunk, int half) {
    ABu b;
    b.q[0] = *(const uint4*)(col_chunk + half * 8);
    b.q[1] = *(const uint4*)(col_chunk + 16 + half * 8);
    return b.v;
}
// A fragment (16x32 bf16): lane holds row=lane&15, K chunks at half*8 and 16+half*8
DEV v16bf loadA_bf16(const __bf16* row_chunk, int half) {
    ABu a;
    a.q[0] = *(const uint4*)(row_chunk + half * 8);
    a.q[1] = *(const uint4*)(row_chunk + 16 + half * 8);
    return a.v;
}
// A fragment from fp32 source, converted with native packed bf16 cvt
DEV v16bf loadA_f32(const float* row_chunk, int half) {
    ABu a;
    const float4 f0 = *(const float4*)(row_chunk + half * 8);
    const float4 f1 = *(const float4*)(row_chunk + half * 8 + 4);
    const float4 f2 = *(const float4*)(row_chunk + 16 + half * 8);
    const float4 f3 = *(const float4*)(row_chunk + 16 + half * 8 + 4);
    a.h[0]=(__bf16)f0.x;  a.h[1]=(__bf16)f0.y;  a.h[2]=(__bf16)f0.z;  a.h[3]=(__bf16)f0.w;
    a.h[4]=(__bf16)f1.x;  a.h[5]=(__bf16)f1.y;  a.h[6]=(__bf16)f1.z;  a.h[7]=(__bf16)f1.w;
    a.h[8]=(__bf16)f2.x;  a.h[9]=(__bf16)f2.y;  a.h[10]=(__bf16)f2.z; a.h[11]=(__bf16)f2.w;
    a.h[12]=(__bf16)f3.x; a.h[13]=(__bf16)f3.y; a.h[14]=(__bf16)f3.z; a.h[15]=(__bf16)f3.w;
    return a.v;
}

// ================= K0: weight transpose + bf16 convert =================
__global__ void k_prep_weights(const float* __restrict__ Wq, const float* __restrict__ Wkv,
                               const float* __restrict__ Wout,
                               __bf16* WqT, __bf16* WkvT, __bf16* WoutT) {
    int idx = blockIdx.x * 256 + threadIdx.x;              // 1,048,576 total
    if (idx < 262144) {
        int dc = idx >> 9, d = idx & 511;
        WqT[idx] = (__bf16)Wq[d * 512 + dc];
    } else if (idx < 786432) {
        int i2 = idx - 262144; int dc = i2 >> 9, d = i2 & 511;
        WkvT[i2] = (__bf16)Wkv[d * 1024 + dc];
    } else if (idx < 1048576) {
        int i3 = idx - 786432; int dc = i3 >> 9, d = i3 & 511;
        WoutT[i3] = (__bf16)Wout[d * 512 + dc];
    }
}

// ================= K1: fused Q / KV projections =================
// q-side: q_tilde[b][m][q][512] = bf16(pre_w[h][m] * scale * (q_in@Wq + bq))
// kv-side: kfull[b][k][512] bf16 ; vT[(b*8+h)*64+d][KP] bf16 (transposed V)
__global__ void k_proj(const float* __restrict__ center, const float* __restrict__ right,
                       const float* __restrict__ summary, const float* __restrict__ memory,
                       const float* __restrict__ bq, const float* __restrict__ bkv,
                       const float* __restrict__ preW,
                       const __bf16* __restrict__ WqT, const __bf16* __restrict__ WkvT,
                       __bf16* qt, __bf16* kf, __bf16* vt) {
    int wave = blockIdx.x * 8 + (threadIdx.x >> 5);
    int lane = threadIdx.x & 31, col = lane & 15, half = lane >> 4;

    if (wave < QTILES * 2 * 32) {                          // ---- q side: 290 row-tiles x 32 d-tiles
        int rt = wave >> 5, dt = wave & 31;
        int rr = rt * 16 + col;                            // A row for this lane
        int q = rr >> 1, b = rr & 1;
        const float* srcrow;
        if (q < RR)            srcrow = right   + (size_t)(q * 2 + b) * DD;
        else if (q < RR + CC)  srcrow = center  + (size_t)((q - RR) * 2 + b) * DD;
        else                   srcrow = summary + (size_t)((q - RR - CC) * 2 + b) * DD;
        const __bf16* wcol = WqT + (size_t)(dt * 16 + col) * DD;

        v8f c = {0.f,0.f,0.f,0.f,0.f,0.f,0.f,0.f};
        #pragma unroll
        for (int cc = 0; cc < 16; ++cc)
            c = wmma_bf16(loadA_f32(srcrow + cc * 32, half), loadB_bf16(wcol + cc * 32, half), c);

        float bias = bq[dt * 16 + col];
        int h = dt >> 2;                                   // head of this 16-col tile
        float pw[8];
        #pragma unroll
        for (int m = 0; m < 8; ++m) pw[m] = preW[h * 8 + m] * 0.125f;  // fold scale = dk^-0.5
        #pragma unroll
        for (int r = 0; r < 8; ++r) {
            int rr2 = rt * 16 + r + half * 8;
            int q2 = rr2 >> 1, b2 = rr2 & 1;
            float val = c[r] + bias;
            #pragma unroll
            for (int m = 0; m < 8; ++m)
                qt[((size_t)(b2 * 8 + m) * QQ + q2) * DD + dt * 16 + col] = (__bf16)(pw[m] * val);
        }
    } else {                                               // ---- kv side: 290 row-tiles x 64 d-tiles
        int t = wave - QTILES * 2 * 32;
        int rt = t >> 6, dt = t & 63;
        int rr = rt * 16 + col;
        int rrc = rr < KK * BB ? rr : 0;                   // clamp pad rows for loads
        int k = rrc >> 1, b = rrc & 1;
        const float* srcrow;
        if (k < MM)            srcrow = memory + (size_t)(k * 2 + b) * DD;
        else if (k < MM + RR)  srcrow = right  + (size_t)((k - MM) * 2 + b) * DD;
        else                   srcrow = center + (size_t)((k - MM - RR) * 2 + b) * DD;
        const __bf16* wcol = WkvT + (size_t)(dt * 16 + col) * DD;

        v8f c = {0.f,0.f,0.f,0.f,0.f,0.f,0.f,0.f};
        #pragma unroll
        for (int cc = 0; cc < 16; ++cc)
            c = wmma_bf16(loadA_f32(srcrow + cc * 32, half), loadB_bf16(wcol + cc * 32, half), c);

        float bias = bkv[dt * 16 + col];
        int dcol = dt * 16 + col;
        #pragma unroll
        for (int r = 0; r < 8; ++r) {
            int rr2 = rt * 16 + r + half * 8;
            if (rr2 >= KK * BB) continue;
            int k2 = rr2 >> 1, b2 = rr2 & 1;
            float val = c[r] + bias;
            if (dcol < DD) {
                kf[((size_t)b2 * KP + k2) * DD + dcol] = (__bf16)val;
            } else {
                int dv = dcol - DD, hh = dv >> 6, din = dv & 63;
                vt[((size_t)((b2 * 8 + hh) * 64 + din)) * KP + k2] = (__bf16)val;
            }
        }
    }
}

DEV int calc_keylen(const int* lengths, int b) {
    int l0 = lengths[0], l1 = lengths[1];
    int mx = l0 > l1 ? l0 : l1;
    int rcb = QQ - mx - SS;
    return (b ? l1 : l0) + MM + rcb;
}

// ================= K2: pass-1 softmax stats (flash-style streaming) =================
__global__ void k_stats(const __bf16* __restrict__ qt, const __bf16* __restrict__ kf,
                        const unsigned char* __restrict__ amask, const int* __restrict__ lengths,
                        float* stats) {
    int wave = blockIdx.x * 8 + (threadIdx.x >> 5);        // 0..2319 = (b, m, qtile)
    int lane = threadIdx.x & 31, col = lane & 15, half = lane >> 4;
    int b = wave / (8 * QTILES);
    int rem = wave % (8 * QTILES);
    int m = rem / QTILES, qtile = rem % QTILES;
    int keylen = calc_keylen(lengths, b);

    // preload invariant A fragments (q_tilde_m tile, full 512 depth)
    const __bf16* arow = qt + ((size_t)(b * 8 + m) * QQ + qtile * 16 + col) * DD;
    ABu afr[16];
    #pragma unroll
    for (int cc = 0; cc < 16; ++cc) {
        afr[cc].q[0] = *(const uint4*)(arow + cc * 32 + half * 8);
        afr[cc].q[1] = *(const uint4*)(arow + cc * 32 + 16 + half * 8);
    }
    float mrun[8], lrun[8];
    #pragma unroll
    for (int r = 0; r < 8; ++r) { mrun[r] = -3.0e38f; lrun[r] = 0.f; }

    const __bf16* kb = kf + (size_t)b * KP * DD;
    for (int kt = 0; kt < KP / 16; ++kt) {
        const __bf16* krow = kb + (size_t)(kt * 16 + col) * DD;
        if (kt + 1 < KP / 16)
            __builtin_prefetch(kb + (size_t)((kt + 1) * 16 + col) * DD, 0, 1);
        v8f c = {0.f,0.f,0.f,0.f,0.f,0.f,0.f,0.f};
        #pragma unroll
        for (int cc = 0; cc < 16; ++cc)
            c = wmma_bf16(afr[cc].v, loadB_bf16(krow + cc * 32, half), c);

        int kc = kt * 16 + col;
        bool pm = kc >= keylen;
        #pragma unroll
        for (int r = 0; r < 8; ++r) {
            int row = qtile * 16 + r + half * 8;
            float w = c[r];
            bool masked = pm || ((kc < KK) && amask[(size_t)row * KK + kc]);
            if (masked) w = MIN_MASK;
            float tmax = w;
            tmax = fmaxf(tmax, __shfl_xor(tmax, 1, 32));
            tmax = fmaxf(tmax, __shfl_xor(tmax, 2, 32));
            tmax = fmaxf(tmax, __shfl_xor(tmax, 4, 32));
            tmax = fmaxf(tmax, __shfl_xor(tmax, 8, 32));
            float mnew = fmaxf(mrun[r], tmax);
            float e = __expf(w - mnew);
            e += __shfl_xor(e, 1, 32);
            e += __shfl_xor(e, 2, 32);
            e += __shfl_xor(e, 4, 32);
            e += __shfl_xor(e, 8, 32);
            lrun[r] = lrun[r] * __expf(mrun[r] - mnew) + e;
            mrun[r] = mnew;
        }
    }
    if (col == 0) {
        #pragma unroll
        for (int r = 0; r < 8; ++r) {
            int row = qtile * 16 + r + half * 8;
            stats[((size_t)(b * 8 + m) * QQ + row) * 2 + 0] = mrun[r];
            stats[((size_t)(b * 8 + m) * QQ + row) * 2 + 1] = lrun[r];
        }
    }
}

// ================= K3: pass-2 attention (P recompute -> LDS -> post-mix -> PV) =================
__global__ void __launch_bounds__(256)
k_attn(const __bf16* __restrict__ qt, const __bf16* __restrict__ kf,
       const __bf16* __restrict__ vt, const unsigned char* __restrict__ amask,
       const int* __restrict__ lengths, const float* __restrict__ stats,
       const float* __restrict__ postW, float* attbuf) {
    __shared__ float LDSP[8][16][36];                      // padded to kill bank conflicts
    int wv = threadIdx.x >> 5;                             // wave id: source head m AND target head n
    int lane = threadIdx.x & 31, col = lane & 15, half = lane >> 4;
    int b = blockIdx.x / QTILES, qtile = blockIdx.x % QTILES;
    int keylen = calc_keylen(lengths, b);

    // final stats for source head m = wv
    float rmax[8], rinv[8];
    #pragma unroll
    for (int r = 0; r < 8; ++r) {
        int row = qtile * 16 + r + half * 8;
        rmax[r] = stats[((size_t)(b * 8 + wv) * QQ + row) * 2 + 0];
        rinv[r] = 1.0f / stats[((size_t)(b * 8 + wv) * QQ + row) * 2 + 1];
    }
    // post-mix column for target head n = wv
    float pw[8];
    #pragma unroll
    for (int mm = 0; mm < 8; ++mm) pw[mm] = postW[mm * 8 + wv];

    // preload invariant A fragments (q_tilde_m tile)
    const __bf16* arow = qt + ((size_t)(b * 8 + wv) * QQ + qtile * 16 + col) * DD;
    ABu afr[16];
    #pragma unroll
    for (int cc = 0; cc < 16; ++cc) {
        afr[cc].q[0] = *(const uint4*)(arow + cc * 32 + half * 8);
        afr[cc].q[1] = *(const uint4*)(arow + cc * 32 + 16 + half * 8);
    }

    v8f attC[4];
    v8f zero = {0.f,0.f,0.f,0.f,0.f,0.f,0.f,0.f};
    #pragma unroll
    for (int d = 0; d < 4; ++d) attC[d] = zero;

    const __bf16* kb = kf + (size_t)b * KP * DD;
    for (int kp = 0; kp < KP / 32; ++kp) {                 // 73 pairs of 16-key tiles
        // ---- stage A: this wave = source head m, produce P_m (16x32) into LDS
        #pragma unroll
        for (int t = 0; t < 2; ++t) {
            int kt = kp * 2 + t;
            const __bf16* krow = kb + (size_t)(kt * 16 + col) * DD;
            v8f c = zero;
            #pragma unroll
            for (int cc = 0; cc < 16; ++cc)
                c = wmma_bf16(afr[cc].v, loadB_bf16(krow + cc * 32, half), c);
            int kc = kt * 16 + col;
            bool pm = kc >= keylen;
            #pragma unroll
            for (int r = 0; r < 8; ++r) {
                int row = qtile * 16 + r + half * 8;
                bool masked = pm || ((kc < KK) && amask[(size_t)row * KK + kc]);
                float p = masked ? 0.f : __expf(c[r] - rmax[r]) * rinv[r];
                LDSP[wv][r + half * 8][t * 16 + col] = p;
            }
        }
        __syncthreads();
        // ---- stage B: this wave = target head n; post-mix probabilities, build A frag
        ABu pa;
        #pragma unroll
        for (int j = 0; j < 8; ++j) {
            int k0 = half * 8 + j, k1 = 16 + half * 8 + j;
            float v0 = 0.f, v1 = 0.f;
            #pragma unroll
            for (int mm = 0; mm < 8; ++mm) {
                v0 += pw[mm] * LDSP[mm][col][k0];
                v1 += pw[mm] * LDSP[mm][col][k1];
            }
            pa.h[j] = (__bf16)v0;
            pa.h[8 + j] = (__bf16)v1;
        }
        // PV: att += P_mixed(16x32) @ V_n(32x16) for 4 d-tiles
        #pragma unroll
        for (int dt = 0; dt < 4; ++dt) {
            const __bf16* vp =
                vt + ((size_t)((b * 8 + wv) * 64 + dt * 16 + col)) * KP + kp * 32 + half * 8;
            ABu bfr;
            bfr.q[0] = *(const uint4*)vp;
            bfr.q[1] = *(const uint4*)(vp + 16);
            attC[dt] = wmma_bf16(pa.v, bfr.v, attC[dt]);
        }
        __syncthreads();
    }
    // write att strip: attbuf[q][b][n*64 + d] fp32
    #pragma unroll
    for (int dt = 0; dt < 4; ++dt)
        #pragma unroll
        for (int r = 0; r < 8; ++r) {
            int q = qtile * 16 + r + half * 8;
            attbuf[((size_t)q * 2 + b) * DD + wv * 64 + dt * 16 + col] = attC[dt][r];
        }
}

// ================= K4: output projection + split/clip epilogue =================
__global__ void k_outproj(const float* __restrict__ attbuf, const __bf16* __restrict__ WoutT,
                          const float* __restrict__ bout, float* out) {
    int wave = blockIdx.x * 8 + (threadIdx.x >> 5);        // 290 row-tiles x 32 d-tiles
    int lane = threadIdx.x & 31, col = lane & 15, half = lane >> 4;
    int rt = wave >> 5, dt = wave & 31;
    int rr = rt * 16 + col;
    const float* arow = attbuf + (size_t)rr * DD;
    const __bf16* wcol = WoutT + (size_t)(dt * 16 + col) * DD;

    v8f c = {0.f,0.f,0.f,0.f,0.f,0.f,0.f,0.f};
    #pragma unroll
    for (int cc = 0; cc < 16; ++cc)
        c = wmma_bf16(loadA_f32(arow + cc * 32, half), loadB_bf16(wcol + cc * 32, half), c);

    float bias = bout[dt * 16 + col];
    int dcol = dt * 16 + col;
    #pragma unroll
    for (int r = 0; r < 8; ++r) {
        int rr2 = rt * 16 + r + half * 8;
        int q = rr2 >> 1, bb = rr2 & 1;
        float val = c[r] + bias;
        if (q < QQ - SS) {
            out[(size_t)q * (BB * DD) + bb * DD + dcol] = val;
        } else {
            int qm = q - (QQ - SS);
            if (qm < SS - 1)
                out[(size_t)(QQ - SS) * (BB * DD) + (size_t)qm * (BB * DD) + bb * DD + dcol] =
                    fminf(fmaxf(val, -10.f), 10.f);
        }
    }
}

// ================= host launcher =================
extern "C" void kernel_launch(void* const* d_in, const int* in_sizes, int n_in,
                              void* d_out, int out_size, void* d_ws, size_t ws_size,
                              hipStream_t stream) {
    const float* center  = (const float*)d_in[0];
    const float* right   = (const float*)d_in[1];
    const int*   lengths = (const int*)d_in[2];
    const float* summary = (const float*)d_in[3];
    const float* memory  = (const float*)d_in[4];
    const unsigned char* amask = (const unsigned char*)d_in[5];
    const float* Wq   = (const float*)d_in[6];
    const float* bq   = (const float*)d_in[7];
    const float* Wkv  = (const float*)d_in[8];
    const float* bkv  = (const float*)d_in[9];
    const float* Wout = (const float*)d_in[10];
    const float* bout = (const float*)d_in[11];
    const float* preW  = (const float*)d_in[12];
    const float* postW = (const float*)d_in[13];
    float* out = (float*)d_out;

    uint8_t* w = (uint8_t*)d_ws;
    __bf16* WqT   = (__bf16*)(w + 0);
    __bf16* WkvT  = (__bf16*)(w + 524288);
    __bf16* WoutT = (__bf16*)(w + 1572864);
    __bf16* qt    = (__bf16*)(w + 2097152);                    // 16*2320*512 bf16
    __bf16* kf    = (__bf16*)(w + 40108032);                   // 2*2336*512  bf16
    __bf16* vt    = (__bf16*)(w + 44892160);                   // 16*64*2336  bf16
    float*  stats = (float*)(w + 49676288);                    // 16*2320*2   f32
    float*  attbuf= (float*)(w + 49973248);                    // 4640*512    f32

    (void)in_sizes; (void)n_in; (void)out_size; (void)ws_size;

    k_prep_weights<<<4096, 256, 0, stream>>>(Wq, Wkv, Wout, WqT, WkvT, WoutT);
    // q side: 290*32 waves, kv side: 290*64 waves -> 27840 waves / 8 = 3480 blocks
    k_proj<<<3480, 256, 0, stream>>>(center, right, summary, memory, bq, bkv, preW,
                                     WqT, WkvT, qt, kf, vt);
    // 2*8*145 = 2320 waves -> 290 blocks
    k_stats<<<290, 256, 0, stream>>>(qt, kf, amask, lengths, stats);
    // one block per (b, qtile) = 290 blocks
    k_attn<<<290, 256, 0, stream>>>(qt, kf, vt, amask, lengths, stats, postW, attbuf);
    // 290*32 waves -> 1160 blocks
    k_outproj<<<1160, 256, 0, stream>>>(attbuf, WoutT, bout, out);
}
